// EmbedderMessageFunction_55997783605364
// MI455X (gfx1250) — matrix operand order repaired and checked by reading, not compiled, not run on hardware
//
#include <hip/hip_runtime.h>
#include <math.h>

// ---------------------------------------------------------------------------
// TGN message-function MLP for MI455X (gfx1250, wave32, WMMA).
//   msg = [emb[src] | emb[dst] | cos(dt*freq+phase) | feat[evi]]  (E x 512, f32)
//   out = relu(msg @ W1 + b1) @ W2 + b2                           (E x 256, f32)
// GEMMs run on v_wmma_f32_16x16x32_bf16 (bf16 in, f32 accumulate); weights are
// converted to transposed bf16 once per launch into workspace.
// ---------------------------------------------------------------------------

typedef __attribute__((ext_vector_type(8)))  __bf16 v8bf;
typedef __attribute__((ext_vector_type(16))) __bf16 v16bf;
typedef __attribute__((ext_vector_type(8)))  float  v8f;

#define E_TILE 64
#define MSGD   512
#define HID    512
#define OUTD   256
#define STRIDE 520   // 512 + 8 bf16 pad -> row stride 1040B = 260 dwords; 16 lanes
                     // doing b128 fragment loads cover all 64 LDS banks once.

__device__ __forceinline__ v16bf cat8(v8bf lo, v8bf hi) {
  return __builtin_shufflevector(lo, hi, 0, 1, 2, 3, 4, 5, 6, 7,
                                         8, 9, 10, 11, 12, 13, 14, 15);
}

// Transpose + f32->bf16 convert both weight matrices into workspace.
// W1t[n][k] = W1[k][n]  (n<512,k<512); W2t[n][k] = W2[k][n] (n<256,k<512).
__global__ void __launch_bounds__(256) prep_weights(const float* __restrict__ W1,
                                                    const float* __restrict__ W2,
                                                    __bf16* __restrict__ W1t,
                                                    __bf16* __restrict__ W2t) {
  int i = blockIdx.x * 256 + threadIdx.x;
  if (i < HID * MSGD) {
    int n = i >> 9, k = i & 511;
    W1t[i] = (__bf16)W1[k * HID + n];
  } else if (i < HID * MSGD + OUTD * HID) {
    int j = i - HID * MSGD;
    int n = j >> 9, k = j & 511;
    W2t[j] = (__bf16)W2[k * OUTD + n];
  }
}

__global__ void __launch_bounds__(256) tgn_msg_mlp(
    const int* __restrict__ src, const int* __restrict__ dst,
    const float* __restrict__ ts, const int* __restrict__ evi,
    const int* __restrict__ idx, const float* __restrict__ node_emb,
    const float* __restrict__ last_update, const float* __restrict__ feats,
    const float* __restrict__ freq, const float* __restrict__ phase,
    const float* __restrict__ b1, const float* __restrict__ b2,
    const __bf16* __restrict__ W1t, const __bf16* __restrict__ W2t,
    float* __restrict__ out, int nE) {
  __shared__ __align__(16) __bf16 sMsg[E_TILE * STRIDE];  // 66,560 B
  __shared__ __align__(16) __bf16 sH[E_TILE * STRIDE];    // 66,560 B
  __shared__ float s_dt[E_TILE];
  __shared__ int   s_src[E_TILE], s_dst[E_TILE], s_evi[E_TILE];

  const int  tid = threadIdx.x;
  const long e0  = (long)blockIdx.x * E_TILE;

  // --- stage per-event scalars (clamped so tail blocks stay in-bounds) ---
  if (tid < E_TILE) {
    long e = e0 + tid;
    if (e > nE - 1) e = nE - 1;
    s_src[tid] = src[e];
    s_dst[tid] = dst[e];
    s_evi[tid] = evi[e];
    s_dt[tid]  = ts[e] - last_update[idx[e]];
  }
  __syncthreads();

  // --- build 64x512 bf16 message tile in LDS (coalesced gathers) ---
  for (int i = tid; i < E_TILE * 128; i += 256) {
    int   r  = i >> 7, c = i & 127;
    float dt = s_dt[r];
    sMsg[r * STRIDE + c]       = (__bf16)node_emb[(long)s_src[r] * 128 + c];
    sMsg[r * STRIDE + 128 + c] = (__bf16)node_emb[(long)s_dst[r] * 128 + c];
    sMsg[r * STRIDE + 256 + c] = (__bf16)cosf(dt * freq[c] + phase[c]);
    sMsg[r * STRIDE + 384 + c] = (__bf16)feats[(long)s_evi[r] * 128 + c];
  }
  __syncthreads();

  const int wave = tid >> 5;
  const int lane = tid & 31;
  const int lrow = lane & 15;   // M (A) or N (B/C) within a 16-wide tile
  const int hl   = lane >> 4;   // lane-half selector
  const int rt   = wave >> 1;   // row tile 0..3 (16 events each)

  // ---------------- GEMM1: h = relu(msg @ W1 + b1) ----------------
  {
    const int cb = (wave & 1) * 16;  // this wave's 16 column tiles of HID
    v8f acc[16] = {};
    for (int ks = 0; ks < 16; ++ks) {
      // A fragment (ISA 16-bit A 16x32 layout): K chunks [k0,k0+8) & [k0+16,k0+24)
      const int     ka = ks * 32 + hl * 8;
      const __bf16* ap = &sMsg[(rt * 16 + lrow) * STRIDE + ka];
      v16bf a = cat8(*(const v8bf*)ap, *(const v8bf*)(ap + 16));
      // B fragment: K chunks [k0,k0+8) & [k0+16,k0+24), matching A
      const int kb = ks * 32 + hl * 8;
#pragma unroll
      for (int c = 0; c < 16; ++c) {
        const __bf16* bp = &W1t[(long)((cb + c) * 16 + lrow) * HID + kb];
        v16bf b = cat8(*(const v8bf*)bp, *(const v8bf*)(bp + 16));
        acc[c] = __builtin_amdgcn_wmma_f32_16x16x32_bf16(
            false, a, false, b, (short)0, acc[c], false, false);
      }
    }
    // bias + relu, write h tile to LDS as bf16 (C layout: N=lrow, M=r+8*hl)
#pragma unroll
    for (int c = 0; c < 16; ++c) {
      int   n    = (cb + c) * 16 + lrow;
      float bias = b1[n];
#pragma unroll
      for (int r = 0; r < 8; ++r) {
        float v = acc[c][r] + bias;
        v = v > 0.f ? v : 0.f;
        sH[(rt * 16 + hl * 8 + r) * STRIDE + n] = (__bf16)v;
      }
    }
  }
  __syncthreads();

  // ---------------- GEMM2: out = h @ W2 + b2 ----------------
  {
    const int cb2 = (wave & 1) * 8;  // this wave's 8 column tiles of OUTD
    v8f acc[8] = {};
    for (int ks = 0; ks < 16; ++ks) {
      const int     ka = ks * 32 + hl * 8;
      const __bf16* ap = &sH[(rt * 16 + lrow) * STRIDE + ka];
      v16bf a = cat8(*(const v8bf*)ap, *(const v8bf*)(ap + 16));
      const int kb = ks * 32 + hl * 8;
#pragma unroll
      for (int c = 0; c < 8; ++c) {
        const __bf16* bp = &W2t[(long)((cb2 + c) * 16 + lrow) * HID + kb];
        v16bf b = cat8(*(const v8bf*)bp, *(const v8bf*)(bp + 16));
        acc[c] = __builtin_amdgcn_wmma_f32_16x16x32_bf16(
            false, a, false, b, (short)0, acc[c], false, false);
      }
    }
#pragma unroll
    for (int c = 0; c < 8; ++c) {
      int   n    = (cb2 + c) * 16 + lrow;
      float bias = b2[n];
#pragma unroll
      for (int r = 0; r < 8; ++r) {
        long row = e0 + rt * 16 + hl * 8 + r;
        if (row < nE) out[row * OUTD + n] = acc[c][r] + bias;
      }
    }
  }
}

extern "C" void kernel_launch(void* const* d_in, const int* in_sizes, int n_in,
                              void* d_out, int out_size, void* d_ws, size_t ws_size,
                              hipStream_t stream) {
  const int*   src   = (const int*)d_in[0];
  const int*   dst   = (const int*)d_in[1];
  const float* ts    = (const float*)d_in[2];
  const int*   evi   = (const int*)d_in[3];
  const int*   idx   = (const int*)d_in[4];
  const float* nemb  = (const float*)d_in[5];
  const float* lupd  = (const float*)d_in[6];
  const float* feat  = (const float*)d_in[7];
  const float* freq  = (const float*)d_in[8];
  const float* phase = (const float*)d_in[9];
  const float* W1    = (const float*)d_in[10];
  const float* b1    = (const float*)d_in[11];
  const float* W2    = (const float*)d_in[12];
  const float* b2    = (const float*)d_in[13];

  __bf16* W1t = (__bf16*)d_ws;            // 512*512 bf16 = 512 KB
  __bf16* W2t = W1t + HID * MSGD;         // 256*512 bf16 = 256 KB

  const int prep_elems = HID * MSGD + OUTD * HID;
  prep_weights<<<(prep_elems + 255) / 256, 256, 0, stream>>>(W1, W2, W1t, W2t);

  const int nE      = in_sizes[0];
  const int nblocks = (nE + E_TILE - 1) / E_TILE;  // 3125 for E=200k
  tgn_msg_mlp<<<nblocks, 256, 0, stream>>>(src, dst, ts, evi, idx, nemb, lupd,
                                           feat, freq, phase, b1, b2, W1t, W2t,
                                           (float*)d_out, nE);
}
